// NetTGCNUncoarsened_76536317215031
// MI455X (gfx1250) — compile-verified, matrix operand order checked by reading, not run on hardware
//
#include <hip/hip_runtime.h>
#include <hip/hip_bf16.h>

// ---------------- problem constants (from setup_inputs) ----------------
constexpr int N1  = 32768;   // nodes, graph 1
constexpr int E1  = 524288;  // edges, graph 1
constexpr int TT  = 29;      // time steps
constexpr int BB  = 4;       // batch
constexpr int ZW1 = TT * BB; // 116 floats per node (Fin=1)
constexpr int KC  = 12;      // Chebyshev order
constexpr int HH  = 15;      // temporal kernel width
constexpr int HP  = 16;      // padded H for WMMA K-dim
constexpr int G1  = 64;
constexpr int G2  = 32;
constexpr int TO  = TT - HH + 1;       // 15
constexpr int N2  = 148;
constexpr int E2  = 2960;
constexpr int ZW2 = G1 * TO * BB;      // 3840
constexpr int K1P = KC * HP;           // 192  (6 WMMA K-steps)
constexpr int K2P = KC * HP * G1;      // 12288 (384 K-steps)
constexpr int M2P = 160;               // mapping rows padded to 10 tiles
constexpr int CH  = 4096;              // node chunk
constexpr int NCH = N1 / CH;           // 8

typedef __attribute__((ext_vector_type(16))) __bf16 v16bf;
typedef __attribute__((ext_vector_type(8)))  float  v8f;

union FragBF {
  v16bf v;
  uint4 q[2];
  unsigned short s[16];
};

__device__ __forceinline__ unsigned short f2bf(float f) {
  unsigned int u = __float_as_uint(f);
  unsigned int r = (u + 0x7FFFu + ((u >> 16) & 1u)) >> 16;
  return (unsigned short)r;
}

// ---------------- elementwise / sparse kernels ----------------
__global__ void k_zero(float* p, int n) {
  int i = blockIdx.x * 256 + threadIdx.x;
  if (i < n) p[i] = 0.f;
}

__global__ void k_deg(const int* ei, const float* w, float* deg, int E) {
  int e = blockIdx.x * 256 + threadIdx.x;
  if (e < E) atomicAdd(&deg[ei[e]], w[e]);
}

__global__ void k_dis(const float* deg, float* dis, int n) {
  int i = blockIdx.x * 256 + threadIdx.x;
  if (i < n) { float d = deg[i]; dis[i] = (d > 0.f) ? 1.0f / sqrtf(d) : 0.f; }
}

__global__ void k_norm(const int* ei, const float* w, const float* dis, float* nrm, int E) {
  int e = blockIdx.x * 256 + threadIdx.x;
  if (e < E) nrm[e] = -dis[ei[e]] * w[e] * dis[ei[E + e]];
}

__global__ void k_buildT0(const float* x, float* tf, unsigned short* tb) {
  int i = blockIdx.x * 256 + threadIdx.x;
  if (i >= N1 * ZW1) return;
  int n = i / ZW1, c = i % ZW1;
  int t = c >> 2, b = c & 3;
  float v = x[((size_t)b * N1 + n) * TT + t];
  tf[i] = v; tb[i] = f2bf(v);
}

__global__ void k_initneg(float* dst, const float* prev, int n) {
  int i = blockIdx.x * 256 + threadIdx.x;
  if (i < n) dst[i] = -prev[i];
}

// edge-parallel Chebyshev hop: zout[row] += scale*norm[e]*zin[col] (float4 lanes)
__global__ void k_prop(const int* ei, const float* nrm, const float* zin, float* zout,
                       int E, int w4, float scale) {
  long tid = (long)blockIdx.x * 256 + threadIdx.x;
  if (tid >= (long)E * w4) return;
  int e = (int)(tid / w4), c = (int)(tid % w4);
  int r = ei[e], col = ei[E + e];
  float4 z = reinterpret_cast<const float4*>(zin)[(size_t)col * w4 + c];
  float s = scale * nrm[e];
  float* o = zout + ((size_t)r * w4 + c) * 4;
  atomicAdd(o + 0, s * z.x); atomicAdd(o + 1, s * z.y);
  atomicAdd(o + 2, s * z.z); atomicAdd(o + 3, s * z.w);
}

__global__ void k_tobf(const float* src, unsigned short* dst, int n) {
  int i = blockIdx.x * 256 + threadIdx.x;
  if (i < n) dst[i] = f2bf(src[i]);
}

__global__ void k_mapbf(const float* mp, unsigned short* mbf) {
  int i = blockIdx.x * 256 + threadIdx.x;
  if (i >= M2P * N1) return;
  int m = i / N1, n = i % N1;
  mbf[i] = (m < N2) ? f2bf(mp[(size_t)m * N1 + n]) : (unsigned short)0;
}

// ---------------- weight packing into lane-ordered WMMA fragments ----------------
// fragment layout per lane L: elems 0..7 = K(base..base+7), 8..15 = K(16+base..), base = (L&16)?8:0
__global__ void k_packB1(const float* W1, unsigned short* bp) {
  int i = blockIdx.x * 256 + threadIdx.x;
  if (i >= (K1P / 32) * (G1 / 16) * 32 * 16) return;       // 12288
  int e = i & 15, lane = (i >> 4) & 31, nt = (i >> 9) & 3, s = i >> 11;
  int base = (lane & 16) ? 8 : 0;
  int kt = s * 32 + ((e < 8) ? base + e : 16 + base + (e - 8));
  int k = kt >> 4, h = kt & 15;
  int g = nt * 16 + (lane & 15);
  bp[i] = (h < HH) ? f2bf(W1[(k * HH + h) * G1 + g]) : (unsigned short)0;
}

__global__ void k_packB2(const float* W2, unsigned short* bp) {
  int i = blockIdx.x * 256 + threadIdx.x;
  if (i >= (K2P / 32) * (G2 / 16) * 32 * 16) return;       // 393216
  int e = i & 15, lane = (i >> 4) & 31, nt = (i >> 9) & 1, s = i >> 10;
  int base = (lane & 16) ? 8 : 0;
  int kt = s * 32 + ((e < 8) ? base + e : 16 + base + (e - 8));
  int k = kt >> 10, h = (kt >> 6) & 15, f = kt & 63;
  int g = nt * 16 + (lane & 15);
  bp[i] = (h < HH) ? f2bf(W2[(((k * HH + h) * G1 + f) * G2) + g]) : (unsigned short)0;
}

// ---------------- WMMA kernel 1: temporal conv (K*H contraction) + ReLU ----------------
// rows = (n,t,b); per 32-wide K-step, each 8-elem fragment half sits in one k-plane with
// consecutive h -> one base pointer + constant-offset loads; h==15 pad zeroed by cndmask.
__global__ void __launch_bounds__(256)
k_conv1(const unsigned short* tbf, const unsigned short* b1p, const float* bias1,
        unsigned short* rh1t, int chunkbase) {
  int wave = blockIdx.x * 8 + (threadIdx.x >> 5);
  int lane = threadIdx.x & 31;
  if (wave >= CH * TO * BB / 16) return;
  int hb = (lane & 16) ? 8 : 0;
  int grow = chunkbase * TO * BB + wave * 16 + (lane & 15);
  int n = grow / (TO * BB), tb = grow % (TO * BB);
  int t = tb >> 2, b = tb & 3;
  const size_t tks = (size_t)N1 * ZW1;
  const unsigned short* abase = tbf + (size_t)n * ZW1 + (size_t)(t + hb) * 4 + b;
  v8f acc[4] = {};
  for (int s = 0; s < K1P / 32; ++s) {
    FragBF a;
    const unsigned short* p0 = abase + (size_t)(2 * s) * tks;  // k = 2s,   h = hb+e
    const unsigned short* p1 = p0 + tks;                       // k = 2s+1, h = hb+e
#pragma unroll
    for (int e = 0; e < 8; ++e) { a.s[e] = p0[e * 4]; a.s[8 + e] = p1[e * 4]; }
    a.s[7]  = hb ? (unsigned short)0 : a.s[7];    // h==15 pad (upper-half lanes only)
    a.s[15] = hb ? (unsigned short)0 : a.s[15];
#pragma unroll
    for (int g = 0; g < 4; ++g) {
      FragBF bb;
      const uint4* pb = reinterpret_cast<const uint4*>(b1p) + (((size_t)(s * 4 + g) * 32 + lane) * 2);
      bb.q[0] = pb[0]; bb.q[1] = pb[1];
      acc[g] = __builtin_amdgcn_wmma_f32_16x16x32_bf16(false, a.v, false, bb.v,
                                                       (short)0, acc[g], false, false);
    }
  }
#pragma unroll
  for (int g = 0; g < 4; ++g) {
    int gg = g * 16 + (lane & 15);
    float bi = bias1[gg];
#pragma unroll
    for (int r = 0; r < 8; ++r) {
      int row = chunkbase * TO * BB + wave * 16 + r + hb;
      int n2 = row / (TO * BB), tb2 = row % (TO * BB);
      float v = acc[g][r] + bi;
      v = v > 0.f ? v : 0.f;
      rh1t[(size_t)(gg * (TO * BB) + tb2) * CH + (n2 - chunkbase)] = f2bf(v);
    }
  }
}

// ---------------- WMMA kernel 2: out2 += mapping[:,chunk] @ relu_h1  ----------------
__global__ void __launch_bounds__(256)
k_gemm_map(const unsigned short* mbf, const unsigned short* rh1t, float* out2, int chunkbase) {
  int wave = blockIdx.x * 8 + (threadIdx.x >> 5);
  int lane = threadIdx.x & 31;
  const int NT = ZW2 / 16;                       // 240
  if (wave >= (M2P / 16) * NT) return;
  int mt = wave / NT, nt = wave % NT;
  int hb = (lane & 16) ? 8 : 0;
  int m = mt * 16 + (lane & 15);
  int cc = nt * 16 + (lane & 15);
  v8f acc;
#pragma unroll
  for (int r = 0; r < 8; ++r)
    acc[r] = out2[(size_t)(mt * 16 + r + hb) * ZW2 + cc];
  const unsigned short* arow = mbf + (size_t)m * N1 + chunkbase;
  const unsigned short* brow = rh1t + (size_t)cc * CH;
  for (int s = 0; s < CH / 32; ++s) {
    FragBF a, bb;
    a.q[0] = *reinterpret_cast<const uint4*>(arow + s * 32 + hb);
    a.q[1] = *reinterpret_cast<const uint4*>(arow + s * 32 + 16 + hb);
    bb.q[0] = *reinterpret_cast<const uint4*>(brow + s * 32 + hb);
    bb.q[1] = *reinterpret_cast<const uint4*>(brow + s * 32 + 16 + hb);
    acc = __builtin_amdgcn_wmma_f32_16x16x32_bf16(false, a.v, false, bb.v,
                                                  (short)0, acc, false, false);
  }
#pragma unroll
  for (int r = 0; r < 8; ++r)
    out2[(size_t)(mt * 16 + r + hb) * ZW2 + cc] = acc[r];
}

__global__ void k_copyT2(const float* out2, float* t2f, unsigned short* t2b) {
  int i = blockIdx.x * 256 + threadIdx.x;
  if (i >= N2 * ZW2) return;
  float v = out2[i]; t2f[i] = v; t2b[i] = f2bf(v);
}

// ---------------- WMMA kernel 3: stage-2 temporal conv (K*H*Fin contraction) ----------------
// decode collapses to k=s>>5, h=(s>>1)&15 (wave-uniform), f0=(s&1)*32+hb; the h==15 pad
// becomes a uniform pointer-select to a zeroed pad buffer -> zero divergence.
__global__ void __launch_bounds__(256)
k_conv2(const unsigned short* t2b, const unsigned short* b2p, const float* bias2,
        const unsigned short* zpad, float* h2) {
  int wave = blockIdx.x * 8 + (threadIdx.x >> 5);
  int lane = threadIdx.x & 31;
  if (wave >= (N2 * BB) / 16) return;            // 37 tiles
  int hb = (lane & 16) ? 8 : 0;
  int row = wave * 16 + (lane & 15);
  int m = row >> 2, b = row & 3;
  const size_t tks = (size_t)N2 * ZW2;
  const size_t mb = (size_t)m * ZW2 + b;
  v8f acc[2] = {};
  for (int s = 0; s < K2P / 32; ++s) {
    int k = s >> 5;
    int h = (s >> 1) & 15;
    int f0 = ((s & 1) << 5) + hb;
    const unsigned short* pk = t2b + (size_t)k * tks + mb + (size_t)h * 4;
    const unsigned short* p0 = (h < HH) ? (pk + (size_t)f0 * (TO * BB)) : zpad;
    const unsigned short* p1 = (h < HH) ? (pk + (size_t)(f0 + 16) * (TO * BB)) : zpad;
    FragBF a;
#pragma unroll
    for (int e = 0; e < 8; ++e) { a.s[e] = p0[e * (TO * BB)]; a.s[8 + e] = p1[e * (TO * BB)]; }
#pragma unroll
    for (int nt = 0; nt < 2; ++nt) {
      FragBF bb;
      const uint4* pb = reinterpret_cast<const uint4*>(b2p) + (((size_t)(s * 2 + nt) * 32 + lane) * 2);
      bb.q[0] = pb[0]; bb.q[1] = pb[1];
      acc[nt] = __builtin_amdgcn_wmma_f32_16x16x32_bf16(false, a.v, false, bb.v,
                                                        (short)0, acc[nt], false, false);
    }
  }
#pragma unroll
  for (int nt = 0; nt < 2; ++nt) {
    int g = nt * 16 + (lane & 15);
    float bi = bias2[g];
#pragma unroll
    for (int r = 0; r < 8; ++r) {
      int row2 = wave * 16 + r + hb;
      int m2 = row2 >> 2, bb2 = row2 & 3;
      h2[(size_t)m2 * (G2 * BB) + g * BB + bb2] = acc[nt][r] + bi;
    }
  }
}

// ---------------- FC + log_softmax ----------------
__global__ void k_fc(const float* h2, const float* fcw, const float* fcb, float* out) {
  __shared__ float lg[24], lse[4];
  int t = threadIdx.x;
  if (t < 24) {
    int i = t / 6, c = t % 6;
    float acc = fcb[c];
    const float* hp = h2 + (size_t)i * (N2 * G2);
    const float* wp = fcw + (size_t)c * (N2 * G2);
    for (int j = 0; j < N2 * G2; ++j) acc += hp[j] * wp[j];
    lg[t] = acc;
  }
  __syncthreads();
  if (t < 4) {
    float mx = lg[t * 6];
    for (int c = 1; c < 6; ++c) mx = fmaxf(mx, lg[t * 6 + c]);
    float s = 0.f;
    for (int c = 0; c < 6; ++c) s += expf(lg[t * 6 + c] - mx);
    lse[t] = mx + logf(s);
  }
  __syncthreads();
  if (t < 24) out[t] = lg[t] - lse[t / 6];
}

// ---------------- host orchestration ----------------
extern "C" void kernel_launch(void* const* d_in, const int* in_sizes, int n_in,
                              void* d_out, int out_size, void* d_ws, size_t ws_size,
                              hipStream_t stream) {
  const float* x   = (const float*)d_in[0];
  const int*   ei1 = (const int*)d_in[1];
  const float* ew1 = (const float*)d_in[2];
  const float* mp  = (const float*)d_in[3];
  const int*   ei2 = (const int*)d_in[4];
  const float* ew2 = (const float*)d_in[5];
  const float* W1  = (const float*)d_in[6];
  const float* b1  = (const float*)d_in[7];
  const float* W2  = (const float*)d_in[8];
  const float* b2  = (const float*)d_in[9];
  const float* fcw = (const float*)d_in[10];
  const float* fcb = (const float*)d_in[11];
  float* out = (float*)d_out;
  (void)in_sizes; (void)n_in; (void)out_size;

  char* w = (char*)d_ws;
  auto alloc = [&](size_t bytes) -> char* {
    char* p = w; w += (bytes + 255) & ~(size_t)255; return p;
  };
  float* deg1 = (float*)alloc((size_t)N1 * 4);
  float* dis1 = (float*)alloc((size_t)N1 * 4);
  float* nr1  = (float*)alloc((size_t)E1 * 4);
  float* deg2 = (float*)alloc((size_t)N2 * 4);
  float* dis2 = (float*)alloc((size_t)N2 * 4);
  float* nr2  = (float*)alloc((size_t)E2 * 4);
  float* tf[3];  for (int i = 0; i < 3; ++i) tf[i]  = (float*)alloc((size_t)N1 * ZW1 * 4);
  unsigned short* tbf  = (unsigned short*)alloc((size_t)KC * N1 * ZW1 * 2);
  unsigned short* b1p  = (unsigned short*)alloc((size_t)12288 * 2);
  unsigned short* mbf  = (unsigned short*)alloc((size_t)M2P * N1 * 2);
  unsigned short* rh1t = (unsigned short*)alloc((size_t)ZW2 * CH * 2);
  float* out2 = (float*)alloc((size_t)M2P * ZW2 * 4);
  float* t2f[3]; for (int i = 0; i < 3; ++i) t2f[i] = (float*)alloc((size_t)N2 * ZW2 * 4);
  unsigned short* t2b = (unsigned short*)alloc((size_t)KC * N2 * ZW2 * 2);
  unsigned short* b2p = (unsigned short*)alloc((size_t)393216 * 2);
  float* h2   = (float*)alloc((size_t)N2 * G2 * BB * 4);
  float* zpad = (float*)alloc(1024);             // zeroed pad source for conv2 h==15 halves
  if ((size_t)(w - (char*)d_ws) > ws_size) return;  // insufficient scratch

  dim3 blk(256);
  auto g1d = [](long n) { return dim3((unsigned)((n + 255) / 256)); };

  // normalization + pad init
  k_zero<<<g1d(N1), blk, 0, stream>>>(deg1, N1);
  k_zero<<<g1d(N2), blk, 0, stream>>>(deg2, N2);
  k_zero<<<g1d((long)M2P * ZW2), blk, 0, stream>>>(out2, M2P * ZW2);
  k_zero<<<dim3(1), blk, 0, stream>>>(zpad, 256);
  k_deg<<<g1d(E1), blk, 0, stream>>>(ei1, ew1, deg1, E1);
  k_deg<<<g1d(E2), blk, 0, stream>>>(ei2, ew2, deg2, E2);
  k_dis<<<g1d(N1), blk, 0, stream>>>(deg1, dis1, N1);
  k_dis<<<g1d(N2), blk, 0, stream>>>(deg2, dis2, N2);
  k_norm<<<g1d(E1), blk, 0, stream>>>(ei1, ew1, dis1, nr1, E1);
  k_norm<<<g1d(E2), blk, 0, stream>>>(ei2, ew2, dis2, nr2, E2);

  // Chebyshev bases, graph 1
  const int nz1 = N1 * ZW1;
  k_buildT0<<<g1d(nz1), blk, 0, stream>>>(x, tf[0], tbf);
  for (int k = 1; k < KC; ++k) {
    float* cur = tf[(k - 1) % 3];
    float* nxt = tf[k % 3];
    if (k == 1) k_zero<<<g1d(nz1), blk, 0, stream>>>(nxt, nz1);
    else        k_initneg<<<g1d(nz1), blk, 0, stream>>>(nxt, tf[(k - 2) % 3], nz1);
    k_prop<<<g1d((long)E1 * (ZW1 / 4)), blk, 0, stream>>>(ei1, nr1, cur, nxt, E1, ZW1 / 4,
                                                          (k == 1) ? 1.f : 2.f);
    k_tobf<<<g1d(nz1), blk, 0, stream>>>(nxt, tbf + (size_t)k * N1 * ZW1, nz1);
  }

  // weight packs + mapping bf16
  k_packB1<<<g1d(12288), blk, 0, stream>>>(W1, b1p);
  k_packB2<<<g1d(393216), blk, 0, stream>>>(W2, b2p);
  k_mapbf<<<g1d((long)M2P * N1), blk, 0, stream>>>(mp, mbf);

  // fused conv1 + ReLU + mapping GEMM, chunked over nodes (sequential accumulation)
  for (int ch = 0; ch < NCH; ++ch) {
    int cb = ch * CH;
    k_conv1<<<dim3(CH * TO * BB / 16 / 8), blk, 0, stream>>>(tbf, b1p, b1, rh1t, cb);
    k_gemm_map<<<dim3((M2P / 16) * (ZW2 / 16) / 8), blk, 0, stream>>>(mbf, rh1t, out2, cb);
  }

  // Chebyshev bases, graph 2
  const int nz2 = N2 * ZW2;
  k_copyT2<<<g1d(nz2), blk, 0, stream>>>(out2, t2f[0], t2b);
  for (int k = 1; k < KC; ++k) {
    float* cur = t2f[(k - 1) % 3];
    float* nxt = t2f[k % 3];
    if (k == 1) k_zero<<<g1d(nz2), blk, 0, stream>>>(nxt, nz2);
    else        k_initneg<<<g1d(nz2), blk, 0, stream>>>(nxt, t2f[(k - 2) % 3], nz2);
    k_prop<<<g1d((long)E2 * (ZW2 / 4)), blk, 0, stream>>>(ei2, nr2, cur, nxt, E2, ZW2 / 4,
                                                          (k == 1) ? 1.f : 2.f);
    k_tobf<<<g1d(nz2), blk, 0, stream>>>(nxt, t2b + (size_t)k * N2 * ZW2, nz2);
  }

  // stage-2 conv + FC + log_softmax
  k_conv2<<<dim3(((N2 * BB / 16) + 7) / 8), blk, 0, stream>>>(t2b, b2p, b2,
                                                              (const unsigned short*)zpad, h2);
  k_fc<<<dim3(1), dim3(64), 0, stream>>>(h2, fcw, fcb, out);
}
